// VSSBlock_81544249082434
// MI455X (gfx1250) — compile-verified, matrix operand order checked by reading, not run on hardware
//
#include <hip/hip_runtime.h>
#include <hip/hip_bf16.h>
#include <cstddef>

// ---------------- problem constants (from reference) ----------------
#define C_CH   64
#define DI_CH  128
#define NST    16
#define RK     6
#define HH     32
#define WW     32
#define DZ     16
#define LTOK   (HH * WW * DZ)        // 16384
#define NPROJ  (2 * NST + 1)         // 33
#define NCHUNK 64                    // chunks per direction for the parallel scan
#define CHLEN  (LTOK / NCHUNK)       // 256 steps per chunk

typedef __attribute__((ext_vector_type(16))) _Float16 v16h;
typedef __attribute__((ext_vector_type(8)))  float    v8f;

// ---------------------------------------------------------------
// K1: fold LoRA into effective weights, convert to f16
//   w_in_eff  (256 x 64)  = in_w  + in_b @ in_a
//   w_out_eff ( 64 x 128) = out_w + out_b @ out_a
// ---------------------------------------------------------------
__global__ void vss_prep_weights(const float* __restrict__ in_w,
                                 const float* __restrict__ in_a,
                                 const float* __restrict__ in_b,
                                 const float* __restrict__ out_w,
                                 const float* __restrict__ out_a,
                                 const float* __restrict__ out_b,
                                 _Float16* __restrict__ w_in,
                                 _Float16* __restrict__ w_out) {
  int t = blockIdx.x * blockDim.x + threadIdx.x;
  if (t < 2 * DI_CH * C_CH) {            // 256*64
    int n = t >> 6, k = t & 63;
    float v = in_w[t];
#pragma unroll
    for (int r = 0; r < RK; ++r) v += in_b[n * RK + r] * in_a[r * C_CH + k];
    w_in[t] = (_Float16)v;
  } else {
    int u = t - 2 * DI_CH * C_CH;
    if (u < C_CH * DI_CH) {              // 64*128
      int c = u >> 7, k = u & 127;
      float v = out_w[u];
#pragma unroll
      for (int r = 0; r < RK; ++r) v += out_b[c * RK + r] * out_a[r * DI_CH + k];
      w_out[u] = (_Float16)v;
    }
  }
}

// ---------------------------------------------------------------
// K2: LayerNorm over C=64 channels; x is (C, L) channel-major.
// One wave (32 lanes) per token; lane holds channels c and c+32.
// ---------------------------------------------------------------
__global__ void vss_layernorm(const float* __restrict__ x,
                              const float* __restrict__ g,
                              const float* __restrict__ b,
                              _Float16* __restrict__ xn) {
  int lane = threadIdx.x & 31;
  int wave = threadIdx.x >> 5;
  int l = blockIdx.x * (blockDim.x >> 5) + wave;
  if (l >= LTOK) return;
  float v0 = x[(size_t)lane * LTOK + l];
  float v1 = x[(size_t)(lane + 32) * LTOK + l];
  float s = v0 + v1;
#pragma unroll
  for (int m = 16; m > 0; m >>= 1) s += __shfl_xor(s, m, 32);
  float mu = s * (1.0f / 64.0f);
  float d0 = v0 - mu, d1 = v1 - mu;
  float vs = d0 * d0 + d1 * d1;
#pragma unroll
  for (int m = 16; m > 0; m >>= 1) vs += __shfl_xor(vs, m, 32);
  float inv = rsqrtf(vs * (1.0f / 64.0f) + 1e-5f);
  xn[(size_t)l * C_CH + lane]      = (_Float16)(d0 * inv * g[lane] + b[lane]);
  xn[(size_t)l * C_CH + lane + 32] = (_Float16)(d1 * inv * g[lane + 32] + b[lane + 32]);
}

// ---------------------------------------------------------------
// K3/K8: WMMA GEMM, all dims compile-time so address math lowers
// to constant-offset addressing.  A: (M x K) f16 row-major.
// Bw: (N x K) f16 (weight layout == B^T).  One wave -> 16x64 slab
// (A frag reused across 4 B frags), K step 32.
//   MODE 0: C[m*N + n] = acc                      (row-major)
//   MODE 1: C[n*M + m] = acc + resid[n*M + m]     (transposed +
//           residual; acc elems i contiguous in m -> float4 x2)
// Fragment layouts per CDNA5 ISA 7.12.2 (lane = r + 16*s):
//   A/B element e: v = e>>3, j = e&7, K = kb + 16*v + 8*s + j
//   acc element i: M = i + 8*s, N = r
// ---------------------------------------------------------------
template <int MODE, int M, int N, int K>
__global__ __launch_bounds__(256)
void vss_wmma_gemm(const _Float16* __restrict__ A,
                   const _Float16* __restrict__ Bw,
                   float* __restrict__ Cout,
                   const float* __restrict__ resid) {
  const int lane = threadIdx.x & 31;
  const int wave = threadIdx.x >> 5;
  constexpr int nGroups = N / 64;
  int gid = blockIdx.x * 8 + wave;                 // 8 waves per block
  if (gid >= (M / 16) * nGroups) return;
  int tm = gid / nGroups;
  int tg = gid - tm * nGroups;
  const int r = lane & 15;
  const int s = lane >> 4;
  const int row = tm * 16 + r;
  const int n0  = tg * 64;

  v8f acc[4] = {};
#pragma unroll 1
  for (int kb = 0; kb < K; kb += 32) {
    v16h a;
    const _Float16* ap = A + (size_t)row * K + kb + 8 * s;
#pragma unroll
    for (int j = 0; j < 8; ++j) { a[j] = ap[j]; a[8 + j] = ap[16 + j]; }
    v16h bf[4];
#pragma unroll
    for (int t = 0; t < 4; ++t) {
      const _Float16* bp = Bw + (size_t)(n0 + t * 16 + r) * K + kb + 8 * s;
#pragma unroll
      for (int j = 0; j < 8; ++j) { bf[t][j] = bp[j]; bf[t][8 + j] = bp[16 + j]; }
    }
#pragma unroll
    for (int t = 0; t < 4; ++t) {
      acc[t] = __builtin_amdgcn_wmma_f32_16x16x32_f16(
          false, a, false, bf[t], (short)0, acc[t], false, false);
    }
  }
#pragma unroll
  for (int t = 0; t < 4; ++t) {
    int col = n0 + t * 16 + r;
    if (MODE == 0) {
      float* cp = Cout + (size_t)(tm * 16 + 8 * s) * N + col;
#pragma unroll
      for (int i = 0; i < 8; ++i) cp[(size_t)i * N] = acc[t][i];
    } else {
      // acc elems i=0..7 map to 8 consecutive rows m -> contiguous
      // along column-major output: two aligned float4 transactions.
      size_t o = (size_t)col * M + (size_t)tm * 16 + 8 * s;
      const float4* rp = (const float4*)(resid + o);
      float4 r0 = rp[0], r1 = rp[1];
      float4 s0 = make_float4(acc[t][0] + r0.x, acc[t][1] + r0.y,
                              acc[t][2] + r0.z, acc[t][3] + r0.w);
      float4 s1 = make_float4(acc[t][4] + r1.x, acc[t][5] + r1.y,
                              acc[t][6] + r1.z, acc[t][7] + r1.w);
      float4* op = (float4*)(Cout + o);
      op[0] = s0;
      op[1] = s1;
    }
  }
}

// ---------------------------------------------------------------
// K4: depthwise 3D conv 4x4x4, pad (2,2,2), crop to (H,W,DZ),
// + bias, SiLU.  xi = xz[:, :128]; out xc (L x 128).
// ---------------------------------------------------------------
__global__ void vss_dwconv_silu(const float* __restrict__ xz,
                                const float* __restrict__ cw,
                                const float* __restrict__ cb,
                                float* __restrict__ xc) {
  int t = blockIdx.x * blockDim.x + threadIdx.x;
  if (t >= LTOK * DI_CH) return;
  int d = t & (DI_CH - 1);
  int l = t >> 7;
  int h  = l / (WW * DZ);
  int rm = l - h * (WW * DZ);
  int w  = rm / DZ;
  int dz = rm - w * DZ;
  float acc = cb[d];
  const float* wd = cw + d * 64;
#pragma unroll
  for (int i = 0; i < 4; ++i) {
    int ih = h - 2 + i;
    if (ih < 0 || ih >= HH) continue;
#pragma unroll
    for (int j = 0; j < 4; ++j) {
      int iw = w - 2 + j;
      if (iw < 0 || iw >= WW) continue;
#pragma unroll
      for (int k = 0; k < 4; ++k) {
        int iz = dz - 2 + k;
        if (iz < 0 || iz >= DZ) continue;
        int lin = (ih * WW + iw) * DZ + iz;
        acc += wd[i * 16 + j * 4 + k] * xz[(size_t)lin * 256 + d];
      }
    }
  }
  xc[t] = acc / (1.0f + __expf(-acc));      // SiLU
}

// ---------------------------------------------------------------
// K5: proj[l, j] = sum_k xc[l,k] * xproj_w[j,k],  j in [0,33)
// ---------------------------------------------------------------
__global__ void vss_xproj(const float* __restrict__ xc,
                          const float* __restrict__ pw,
                          float* __restrict__ proj) {
  int t = blockIdx.x * blockDim.x + threadIdx.x;
  if (t >= LTOK * NPROJ) return;
  int l = t / NPROJ;
  int j = t - l * NPROJ;
  const float* xr = xc + (size_t)l * DI_CH;
  const float* wr = pw + (size_t)j * DI_CH;
  float acc = 0.f;
#pragma unroll 8
  for (int k = 0; k < DI_CH; ++k) acc += xr[k] * wr[k];
  proj[t] = acc;
}

// ---------------------------------------------------------------
// K6: dt[l,d] = softplus(proj[l,0]*dt_w[d] + dt_b[d])
// ---------------------------------------------------------------
__global__ void vss_dt(const float* __restrict__ proj,
                       const float* __restrict__ dtw,
                       const float* __restrict__ dtb,
                       float* __restrict__ dt) {
  int t = blockIdx.x * blockDim.x + threadIdx.x;
  if (t >= LTOK * DI_CH) return;
  int d = t & (DI_CH - 1);
  int l = t >> 7;
  float v = proj[(size_t)l * NPROJ] * dtw[d] + dtb[d];
  dt[t] = (v > 20.f) ? v : log1pf(__expf(v));
}

// ---------------------------------------------------------------
// K7: zero ys accumulator + mean slot per launch (harness poisons
// once and never re-poisons).
// ---------------------------------------------------------------
__global__ void vss_zero(float* __restrict__ ys, float* __restrict__ ymean, int n) {
  int t = blockIdx.x * blockDim.x + threadIdx.x;
  if (t < n) ys[t] = 0.f;
  if (t < DI_CH) ymean[t] = 0.f;
}

// ---------------------------------------------------------------
// Scan position -> token index mapping per direction.
// dir0: identity; dir1: reversed; dir2: (dz,h,w) order; dir3: rev.
// reorder_back composes to identity, so outputs land at token idx.
// ---------------------------------------------------------------
__device__ __forceinline__ int vss_map(int dir, int pos) {
  int p = (dir & 1) ? (LTOK - 1 - pos) : pos;
  if (dir >= 2) {
    int dz  = p >> 10;              // p / (H*W)
    int rem = p & 1023;             // h*W + w
    return rem * DZ + dz;
  }
  return p;
}

// ---------------------------------------------------------------
// Chunked parallel scan. The recurrence h <- dA*h + dBx is affine
// per (d,n), so chunks compose:
//  P1: per (dir,chunk,d): local h (zero init) + decay product P
//  P2: per (dir,d): tiny sequential pass over 64 chunk summaries
//  P3: per (dir,chunk,d): replay with true h_init, emit y into ys
// 256 blocks x 128 thr for P1/P3 (vs 4 blocks for a naive scan).
// ---------------------------------------------------------------
__global__ __launch_bounds__(DI_CH)
void vss_scan_phase1(const float* __restrict__ dt,
                     const float* __restrict__ proj,
                     const float* __restrict__ xc,
                     const float* __restrict__ a_log,
                     float* __restrict__ hloc,
                     float* __restrict__ pprod) {
  const int d     = threadIdx.x;
  const int chunk = blockIdx.x & (NCHUNK - 1);
  const int dir   = blockIdx.x >> 6;          // NCHUNK == 64
  float Aneg[NST];
#pragma unroll
  for (int n = 0; n < NST; ++n) Aneg[n] = -__expf(a_log[d * NST + n]);
  float h[NST], P[NST];
#pragma unroll
  for (int n = 0; n < NST; ++n) { h[n] = 0.f; P[n] = 1.f; }
  const int base = chunk * CHLEN;
  for (int q = 0; q < CHLEN; ++q) {
    int idx = vss_map(dir, base + q);
    float dtv = dt[(size_t)idx * DI_CH + d];
    float xv  = xc[(size_t)idx * DI_CH + d];
    const float* pj = proj + (size_t)idx * NPROJ;
#pragma unroll
    for (int n = 0; n < NST; ++n) {
      float dA = __expf(Aneg[n] * dtv);
      P[n] *= dA;
      h[n] = dA * h[n] + dtv * pj[1 + n] * xv;
    }
  }
  size_t o = (((size_t)dir * NCHUNK + chunk) * DI_CH + d) * NST;
#pragma unroll
  for (int n = 0; n < NST; ++n) { hloc[o + n] = h[n]; pprod[o + n] = P[n]; }
}

__global__ void vss_scan_phase2(const float* __restrict__ hloc,
                                const float* __restrict__ pprod,
                                float* __restrict__ hstart) {
  int t = blockIdx.x * blockDim.x + threadIdx.x;   // dir*128 + d
  if (t >= 4 * DI_CH) return;
  int dir = t >> 7;
  int d   = t & (DI_CH - 1);
  float run[NST];
#pragma unroll
  for (int n = 0; n < NST; ++n) run[n] = 0.f;
  for (int c = 0; c < NCHUNK; ++c) {
    size_t o = (((size_t)dir * NCHUNK + c) * DI_CH + d) * NST;
#pragma unroll
    for (int n = 0; n < NST; ++n) {
      float rn = run[n];
      hstart[o + n] = rn;
      run[n] = pprod[o + n] * rn + hloc[o + n];
    }
  }
}

__global__ __launch_bounds__(DI_CH)
void vss_scan_phase3(const float* __restrict__ dt,
                     const float* __restrict__ proj,
                     const float* __restrict__ xc,
                     const float* __restrict__ a_log,
                     const float* __restrict__ hstart,
                     float* __restrict__ ys) {
  const int d     = threadIdx.x;
  const int chunk = blockIdx.x & (NCHUNK - 1);
  const int dir   = blockIdx.x >> 6;
  float Aneg[NST];
#pragma unroll
  for (int n = 0; n < NST; ++n) Aneg[n] = -__expf(a_log[d * NST + n]);
  float h[NST];
  {
    size_t o = (((size_t)dir * NCHUNK + chunk) * DI_CH + d) * NST;
#pragma unroll
    for (int n = 0; n < NST; ++n) h[n] = hstart[o + n];
  }
  const int base = chunk * CHLEN;
  for (int q = 0; q < CHLEN; ++q) {
    int idx = vss_map(dir, base + q);
    if (q + 4 < CHLEN) {
      int in2 = vss_map(dir, base + q + 4);
      __builtin_prefetch(dt + (size_t)in2 * DI_CH + d, 0, 1);
      __builtin_prefetch(xc + (size_t)in2 * DI_CH + d, 0, 1);
    }
    float dtv = dt[(size_t)idx * DI_CH + d];
    float xv  = xc[(size_t)idx * DI_CH + d];
    const float* pj = proj + (size_t)idx * NPROJ;
    float acc = 0.f;
#pragma unroll
    for (int n = 0; n < NST; ++n) {
      float dA = __expf(Aneg[n] * dtv);
      h[n] = dA * h[n] + dtv * pj[1 + n] * xv;
      acc += h[n] * pj[17 + n];
    }
    atomicAdd(&ys[(size_t)idx * DI_CH + d], acc);
  }
}

// ---------------------------------------------------------------
// K10: y = (ys/4 + xc*d_param) * silu(z); emit f16 for out-GEMM,
// accumulate channel mean into d_out tail.
// ---------------------------------------------------------------
__global__ void vss_combine(const float* __restrict__ ys,
                            const float* __restrict__ xc,
                            const float* __restrict__ xz,
                            const float* __restrict__ dparam,
                            _Float16* __restrict__ yh,
                            float* __restrict__ ymean) {
  int t = blockIdx.x * blockDim.x + threadIdx.x;
  if (t >= LTOK * DI_CH) return;
  int d = t & (DI_CH - 1);
  int l = t >> 7;
  float z  = xz[(size_t)l * 256 + DI_CH + d];
  float sz = z / (1.0f + __expf(-z));
  float yv = (ys[t] * 0.25f + xc[t] * dparam[d]) * sz;
  yh[t] = (_Float16)yv;
  atomicAdd(&ymean[d], yv * (1.0f / (float)LTOK));
}

// ---------------------------------------------------------------
extern "C" void kernel_launch(void* const* d_in, const int* in_sizes, int n_in,
                              void* d_out, int out_size, void* d_ws, size_t ws_size,
                              hipStream_t stream) {
  const float* x       = (const float*)d_in[0];
  const float* norm_g  = (const float*)d_in[1];
  const float* norm_b  = (const float*)d_in[2];
  const float* in_w    = (const float*)d_in[3];
  const float* in_a    = (const float*)d_in[4];
  const float* in_b    = (const float*)d_in[5];
  const float* conv_w  = (const float*)d_in[6];
  const float* conv_b  = (const float*)d_in[7];
  const float* a_log   = (const float*)d_in[8];
  const float* d_param = (const float*)d_in[9];
  const float* xproj_w = (const float*)d_in[10];
  const float* dt_w    = (const float*)d_in[11];
  const float* dt_b    = (const float*)d_in[12];
  const float* out_w   = (const float*)d_in[13];
  const float* out_a   = (const float*)d_in[14];
  const float* out_b   = (const float*)d_in[15];

  float* out   = (float*)d_out;
  float* ymean = out + (size_t)C_CH * LTOK;   // second tuple output (128 floats)

  // ---- workspace carve (256B aligned) ----
  char* ws = (char*)d_ws;
  size_t off = 0;
  auto carve = [&](size_t bytes) { size_t o = off; off += (bytes + 255) & ~(size_t)255; return o; };
  _Float16* xn_h  = (_Float16*)(ws + carve((size_t)LTOK * C_CH * 2));          // 2 MB
  _Float16* win_h = (_Float16*)(ws + carve((size_t)2 * DI_CH * C_CH * 2));     // 32 KB
  _Float16* wout_h= (_Float16*)(ws + carve((size_t)C_CH * DI_CH * 2));         // 16 KB
  float*    xz    = (float*)   (ws + carve((size_t)LTOK * 256 * 4));           // 16 MB
  float*    xc    = (float*)   (ws + carve((size_t)LTOK * DI_CH * 4));         // 8 MB
  float*    proj  = (float*)   (ws + carve((size_t)LTOK * NPROJ * 4));         // ~2 MB
  float*    dt    = (float*)   (ws + carve((size_t)LTOK * DI_CH * 4));         // 8 MB
  float*    ys    = (float*)   (ws + carve((size_t)LTOK * DI_CH * 4));         // 8 MB
  _Float16* y_h   = (_Float16*)(ws + carve((size_t)LTOK * DI_CH * 2));         // 4 MB
  float*    hloc  = (float*)   (ws + carve((size_t)4 * NCHUNK * DI_CH * NST * 4)); // 2 MB
  float*    pprod = (float*)   (ws + carve((size_t)4 * NCHUNK * DI_CH * NST * 4)); // 2 MB
  float*    hstart= (float*)   (ws + carve((size_t)4 * NCHUNK * DI_CH * NST * 4)); // 2 MB
  (void)ws_size; (void)in_sizes; (void)n_in; (void)out_size;

  const int TB = 256;

  // K1: effective weights (folded LoRA), f16
  vss_prep_weights<<<(2 * DI_CH * C_CH + C_CH * DI_CH + TB - 1) / TB, TB, 0, stream>>>(
      in_w, in_a, in_b, out_w, out_a, out_b, win_h, wout_h);

  // K2: layernorm -> xn (L x 64) f16
  vss_layernorm<<<LTOK / 8, TB, 0, stream>>>(x, norm_g, norm_b, xn_h);

  // K3: input GEMM (L x 64) @ (256 x 64)^T -> xz (L x 256), WMMA
  {
    constexpr int tiles = (LTOK / 16) * (256 / 64);       // 4096 waves
    vss_wmma_gemm<0, LTOK, 256, C_CH><<<tiles / 8, TB, 0, stream>>>(
        xn_h, win_h, xz, nullptr);
  }

  // K4: depthwise conv + SiLU -> xc
  vss_dwconv_silu<<<(LTOK * DI_CH) / TB, TB, 0, stream>>>(xz, conv_w, conv_b, xc);

  // K5: x-projection -> proj (L x 33)
  vss_xproj<<<(LTOK * NPROJ + TB - 1) / TB, TB, 0, stream>>>(xc, xproj_w, proj);

  // K6: dt = softplus(...)
  vss_dt<<<(LTOK * DI_CH) / TB, TB, 0, stream>>>(proj, dt_w, dt_b, dt);

  // K7: zero scan accumulator + mean slot
  vss_zero<<<(LTOK * DI_CH) / TB, TB, 0, stream>>>(ys, ymean, LTOK * DI_CH);

  // K9: chunked parallel scan (4 dirs x 64 chunks)
  vss_scan_phase1<<<4 * NCHUNK, DI_CH, 0, stream>>>(dt, proj, xc, a_log, hloc, pprod);
  vss_scan_phase2<<<2, TB, 0, stream>>>(hloc, pprod, hstart);
  vss_scan_phase3<<<4 * NCHUNK, DI_CH, 0, stream>>>(dt, proj, xc, a_log, hstart, ys);

  // K10: gate, combine, mean
  vss_combine<<<(LTOK * DI_CH) / TB, TB, 0, stream>>>(ys, xc, xz, d_param, y_h, ymean);

  // K8: output GEMM (L x 128) @ (64 x 128)^T, transposed store + residual
  {
    constexpr int tiles = (LTOK / 16) * (C_CH / 64);      // 1024 waves
    vss_wmma_gemm<1, LTOK, C_CH, DI_CH><<<tiles / 8, TB, 0, stream>>>(
        y_h, wout_h, out, x);
  }
}
